// HRM_43920335568907
// MI455X (gfx1250) — compile-verified, matrix operand order checked by reading, not run on hardware
//
#include <hip/hip_runtime.h>
#include <math.h>

#define NNODE 8192
#define NMELT (NNODE * 128)

typedef __attribute__((ext_vector_type(16))) _Float16 v16h;
typedef __attribute__((ext_vector_type(8)))  _Float16 v8h;
typedef __attribute__((ext_vector_type(8)))  float    v8f;

__device__ __forceinline__ float geluf(float x) {
    return 0.5f * x * (1.0f + erff(x * 0.70710678118654752f));
}
__device__ __forceinline__ float lreluf(float x) { return x >= 0.f ? x : 0.2f * x; }
__device__ __forceinline__ unsigned f2ord(float f) {
    unsigned u = __float_as_uint(f);
    return (u & 0x80000000u) ? ~u : (u | 0x80000000u);
}
__device__ __forceinline__ float ord2f(unsigned u) {
    return (u & 0x80000000u) ? __uint_as_float(u & 0x7fffffffu) : __uint_as_float(~u);
}

// block of exactly 128 threads: returns total sum to all threads
__device__ __forceinline__ float blockSum128(float v) {
    __shared__ float red[4];
    __shared__ float tot;
    __syncthreads();
#pragma unroll
    for (int off = 16; off; off >>= 1) v += __shfl_down(v, off);
    if ((threadIdx.x & 31) == 0) red[threadIdx.x >> 5] = v;
    __syncthreads();
    if (threadIdx.x == 0) tot = red[0] + red[1] + red[2] + red[3];
    __syncthreads();
    return tot;
}

// ---------------------------------------------------------------------------
// WMMA GEMM: C[M,N] = act(A[M,K] @ B[K,N] + bias), f32 in/out, f16 WMMA math.
// Requires M%16==0, N%32==0, K%32==0. Block = 128 threads = 4 waves.
// Block tile = 64 rows x 32 cols: each wave owns 16 rows x 32 cols
// (2 accumulators sharing one A fragment). B tile (32k x 32n) is staged once
// per block into LDS as f16 transposed [n][k] so each lane's WMMA fragment is
// two contiguous 16B ds loads. grid.z batches via element strides aB/bB/cB.
// ---------------------------------------------------------------------------
__global__ void k_gemm(const float* __restrict__ A, const float* __restrict__ Bw,
                       const float* __restrict__ bias, float* __restrict__ C,
                       int M, int N, int K, int lda, int ldb, int ldc,
                       long aB, long bB, long cB, int act) {
    A  += (long)blockIdx.z * aB;
    Bw += (long)blockIdx.z * bB;
    C  += (long)blockIdx.z * cB;
    __shared__ _Float16 Bs[32 * 32];  // [n][k] transposed tile
    int lane  = threadIdx.x & 31;
    int wave  = threadIdx.x >> 5;
    int tileN = blockIdx.x * 32;
    int tileM = (blockIdx.y * 4 + wave) * 16;
    bool mAct = tileM < M;            // inactive waves still join barriers
    int half = lane >> 4;             // K-half selector per WMMA 16-bit layout
    int l16  = lane & 15;
    const float* ap0 = A + (long)((mAct ? tileM : 0) + l16) * lda;
    // B staging assignment: thread -> 4 n-values x 2 k-values
    int sn = (threadIdx.x & 7) * 4;
    int sk = threadIdx.x >> 3;        // 0..15
    v8f acc0 = {}, acc1 = {};
    for (int kk = 0; kk < K; kk += 32) {
        if (kk + 32 < K) __builtin_prefetch(ap0 + kk + 32, 0, 0);
        __syncthreads();
#pragma unroll
        for (int it = 0; it < 2; ++it) {
            int k = sk + it * 16;
            float4 bv = *(const float4*)(Bw + (long)(kk + k) * ldb + tileN + sn);
            Bs[(sn + 0) * 32 + k] = (_Float16)bv.x;
            Bs[(sn + 1) * 32 + k] = (_Float16)bv.y;
            Bs[(sn + 2) * 32 + k] = (_Float16)bv.z;
            Bs[(sn + 3) * 32 + k] = (_Float16)bv.w;
        }
        __syncthreads();
        // A fragment: two contiguous 8-float chunks -> f16
        v16h af;
        const float* ap = ap0 + kk + half * 8;
        float4 a0 = ((const float4*)ap)[0];
        float4 a1 = ((const float4*)ap)[1];
        float4 a2 = ((const float4*)(ap + 16))[0];
        float4 a3 = ((const float4*)(ap + 16))[1];
        af[0] = (_Float16)a0.x; af[1] = (_Float16)a0.y; af[2] = (_Float16)a0.z; af[3] = (_Float16)a0.w;
        af[4] = (_Float16)a1.x; af[5] = (_Float16)a1.y; af[6] = (_Float16)a1.z; af[7] = (_Float16)a1.w;
        af[8] = (_Float16)a2.x; af[9] = (_Float16)a2.y; af[10] = (_Float16)a2.z; af[11] = (_Float16)a2.w;
        af[12] = (_Float16)a3.x; af[13] = (_Float16)a3.y; af[14] = (_Float16)a3.z; af[15] = (_Float16)a3.w;
        // B fragments from LDS (two 16-col subtiles)
        v16h bf0, bf1;
        {
            const _Float16* b0 = &Bs[(long)l16 * 32 + half * 8];
            v8h lo = *(const v8h*)b0;
            v8h hi = *(const v8h*)(b0 + 16);
#pragma unroll
            for (int j = 0; j < 8; ++j) { bf0[j] = lo[j]; bf0[8 + j] = hi[j]; }
            const _Float16* b1 = &Bs[(long)(16 + l16) * 32 + half * 8];
            v8h lo1 = *(const v8h*)b1;
            v8h hi1 = *(const v8h*)(b1 + 16);
#pragma unroll
            for (int j = 0; j < 8; ++j) { bf1[j] = lo1[j]; bf1[8 + j] = hi1[j]; }
        }
        acc0 = __builtin_amdgcn_wmma_f32_16x16x32_f16(false, af, false, bf0,
                                                      (short)0, acc0, false, false);
        acc1 = __builtin_amdgcn_wmma_f32_16x16x32_f16(false, af, false, bf1,
                                                      (short)0, acc1, false, false);
    }
    if (mAct) {
        int col0 = tileN + l16;
        float bv0 = bias ? bias[col0] : 0.0f;
        float bv1 = bias ? bias[col0 + 16] : 0.0f;
#pragma unroll
        for (int r = 0; r < 8; ++r) {
            int row = tileM + r + half * 8;
            float v0 = acc0[r] + bv0;
            float v1 = acc1[r] + bv1;
            if (act == 1) { v0 = geluf(v0); v1 = geluf(v1); }
            C[(long)row * ldc + col0]      = v0;
            C[(long)row * ldc + col0 + 16] = v1;
        }
    }
}

// ---------------------------------------------------------------------------
// elementwise / fill kernels
// ---------------------------------------------------------------------------
__global__ void k_fill_f32(float* p, long n, float v) {
    long i = (long)blockIdx.x * 256 + threadIdx.x;
    if (i < n) p[i] = v;
}
__global__ void k_fill_u32(unsigned* p, long n, unsigned v) {
    long i = (long)blockIdx.x * 256 + threadIdx.x;
    if (i < n) p[i] = v;
}
__global__ void k_rowbcast(float* out, const float* __restrict__ row, long n) {
    long i = (long)blockIdx.x * 256 + threadIdx.x;
    if (i < n) out[i] = row[i & 127];
}
__global__ void k_gelu_ip(float* p, long n) {
    long i = (long)blockIdx.x * 256 + threadIdx.x;
    if (i < n) p[i] = geluf(p[i]);
}
__global__ void k_accinit3(float* acc, const float* __restrict__ b0,
                           const float* __restrict__ b1, const float* __restrict__ b2) {
    long i = (long)blockIdx.x * 256 + threadIdx.x;
    if (i < NMELT) { int d = (int)(i & 127); acc[i] = b0[d] + b1[d] + b2[d]; }
}

// emb = rmsnorm(gelu(x @ We + be), scale); K=16, block=128 (one node per block)
__global__ void k_embed(const float* __restrict__ x, const float* __restrict__ W,
                        const float* __restrict__ b, const float* __restrict__ sc,
                        float* __restrict__ out) {
    int n = blockIdx.x, d = threadIdx.x;
    const float* xr = x + n * 16;
    float a = b[d];
#pragma unroll
    for (int k = 0; k < 16; ++k) a += xr[k] * W[k * 128 + d];
    float g  = geluf(a);
    float ss = blockSum128(g * g);
    out[(long)n * 128 + d] = sc[d] * g * rsqrtf(ss * (1.f / 128.f) + 1e-6f);
}

// GAT attention coefficients: as_[t,n,h], ad_[t,n,h]; H=2, Dh=64
__global__ void k_gat_coef(const float* __restrict__ h3, const float* __restrict__ asrc,
                           const float* __restrict__ adst, float* __restrict__ as_,
                           float* __restrict__ ad_) {
    int n = blockIdx.x, t = blockIdx.y, d = threadIdx.x;
    const float* h = h3 + (long)t * NMELT + (long)n * 128;
    __shared__ float sa[128], sb[128];
    float hv = h[d];
    sa[d] = hv * asrc[t * 128 + d];
    sb[d] = hv * adst[t * 128 + d];
    __syncthreads();
    if (d < 2) {
        float xs = 0.f, ys = 0.f;
        for (int j = 0; j < 64; ++j) { xs += sa[d * 64 + j]; ys += sb[d * 64 + j]; }
        as_[((long)t * NNODE + n) * 2 + d] = xs;
        ad_[((long)t * NNODE + n) * 2 + d] = ys;
    }
}

// GAT edge pass A: segment max (ordered-uint encoding). Edges + self loops.
__global__ void k_gat_eA(const int* __restrict__ ei, int E,
                         const float* __restrict__ as_, const float* __restrict__ ad_,
                         unsigned* __restrict__ menc) {
    int e = blockIdx.x * 256 + threadIdx.x;
    int tot = E + NNODE;
    if (e >= tot) return;
    int s, d;
    if (e < E) { s = ei[e]; d = ei[E + e]; } else { s = d = e - E; }
#pragma unroll
    for (int h = 0; h < 2; ++h) {
        float v = lreluf(as_[(long)s * 2 + h] + ad_[(long)d * 2 + h]);
        atomicMax(&menc[(long)d * 2 + h], f2ord(v));
    }
}

// GAT edge pass B: ex = exp(e - max); store ex; atomic sum per dst
__global__ void k_gat_eB(const int* __restrict__ ei, int E,
                         const float* __restrict__ as_, const float* __restrict__ ad_,
                         const unsigned* __restrict__ menc,
                         float* __restrict__ exb, float* __restrict__ sg) {
    int e = blockIdx.x * 256 + threadIdx.x;
    int tot = E + NNODE;
    if (e >= tot) return;
    int s, d;
    if (e < E) { s = ei[e]; d = ei[E + e]; } else { s = d = e - E; }
#pragma unroll
    for (int h = 0; h < 2; ++h) {
        float v  = lreluf(as_[(long)s * 2 + h] + ad_[(long)d * 2 + h]);
        float ex = expf(v - ord2f(menc[(long)d * 2 + h]));
        exb[(long)e * 2 + h] = ex;
        atomicAdd(&sg[(long)d * 2 + h], ex);
    }
}

// GAT edge pass C: acc[dst] += alpha * h[src]; one wave32 per edge
__global__ void k_gat_eC(const int* __restrict__ ei, int E,
                         const float* __restrict__ h, const float* __restrict__ exb,
                         const float* __restrict__ sg, float* __restrict__ acc) {
    int w = (blockIdx.x * 256 + threadIdx.x) >> 5;
    int lane = threadIdx.x & 31;
    int tot = E + NNODE;
    if (w >= tot) return;
    int s, d;
    if (w < E) { s = ei[w]; d = ei[E + w]; } else { s = d = w - E; }
#pragma unroll
    for (int hh = 0; hh < 2; ++hh) {
        float alpha = exb[(long)w * 2 + hh] / sg[(long)d * 2 + hh];
#pragma unroll
        for (int j = 0; j < 2; ++j) {
            int c = hh * 64 + lane + j * 32;
            atomicAdd(&acc[(long)d * 128 + c], alpha * h[(long)s * 128 + c]);
        }
    }
}

// out = lrelu(prev + rmsnorm(acc, scale))
__global__ void k_combine(const float* __restrict__ prev, const float* __restrict__ acc,
                          const float* __restrict__ sc, float* __restrict__ out) {
    int n = blockIdx.x, d = threadIdx.x;
    float a  = acc[(long)n * 128 + d];
    float ss = blockSum128(a * a);
    float v  = prev[(long)n * 128 + d] + sc[d] * a * rsqrtf(ss * (1.f / 128.f) + 1e-6f);
    out[(long)n * 128 + d] = lreluf(v);
}

// HGT: x3 = lrelu(x2 + rmsnorm(a*o + (1-a)*x2, scale))
__global__ void k_mixcomb(const float* __restrict__ x2, const float* __restrict__ o,
                          const float* __restrict__ skip, const float* __restrict__ sc,
                          float* __restrict__ out) {
    int n = blockIdx.x, d = threadIdx.x;
    float a   = 1.f / (1.f + expf(-skip[0]));
    float mix = a * o[(long)n * 128 + d] + (1.f - a) * x2[(long)n * 128 + d];
    float ss  = blockSum128(mix * mix);
    float v   = x2[(long)n * 128 + d] + sc[d] * mix * rsqrtf(ss * (1.f / 128.f) + 1e-6f);
    out[(long)n * 128 + d] = lreluf(v);
}

// HGT scores: score[e,h] = dot(q[dst,h], kp[src,h]) * prel[h] * 1/sqrt(32); + segment max
__global__ void k_hgt_score(const int* __restrict__ ei, int E, const float* __restrict__ q,
                            const float* __restrict__ kp, const float* __restrict__ prel,
                            float* __restrict__ score, unsigned* __restrict__ mh) {
    int e = blockIdx.x * 256 + threadIdx.x;
    if (e >= E) return;
    int s = ei[e], d = ei[E + e];
#pragma unroll
    for (int h = 0; h < 4; ++h) {
        const float* qp = q + (long)d * 128 + h * 32;
        const float* kq = kp + (long)s * 128 + h * 32;
        float dot = 0.f;
#pragma unroll 8
        for (int f = 0; f < 32; ++f) dot += qp[f] * kq[f];
        float sc = dot * prel[h] * 0.17677669529663687f;
        score[(long)e * 4 + h] = sc;
        atomicMax(&mh[(long)d * 4 + h], f2ord(sc));
    }
}

__global__ void k_hgt_exp(const int* __restrict__ ei, int E, const float* __restrict__ score,
                          const unsigned* __restrict__ mh, float* __restrict__ ex,
                          float* __restrict__ sh) {
    int e = blockIdx.x * 256 + threadIdx.x;
    if (e >= E) return;
    int d = ei[E + e];
#pragma unroll
    for (int h = 0; h < 4; ++h) {
        float v = expf(score[(long)e * 4 + h] - ord2f(mh[(long)d * 4 + h]));
        ex[(long)e * 4 + h] = v;
        atomicAdd(&sh[(long)d * 4 + h], v);
    }
}

// agg[dst,h,:] += alpha * vp[src,h,:]; one wave32 per edge (lane = Dh index)
__global__ void k_hgt_agg(const int* __restrict__ ei, int E, const float* __restrict__ vp,
                          const float* __restrict__ ex, const float* __restrict__ sh,
                          float* __restrict__ agg) {
    int w = (blockIdx.x * 256 + threadIdx.x) >> 5;
    int lane = threadIdx.x & 31;
    if (w >= E) return;
    int s = ei[w], d = ei[E + w];
#pragma unroll
    for (int h = 0; h < 4; ++h) {
        float a = ex[(long)w * 4 + h] / sh[(long)d * 4 + h];
        atomicAdd(&agg[(long)d * 128 + h * 32 + lane],
                  a * vp[(long)s * 128 + h * 32 + lane]);
    }
}

// FiLM: nodes = (1+0.1*tanh(gamma))*(x3+inj) + 0.1*tanh(beta)
__global__ void k_film(const float* __restrict__ x3, const float* __restrict__ inj,
                       const float* __restrict__ gb, float* __restrict__ nodes) {
    long i = (long)blockIdx.x * 256 + threadIdx.x;
    if (i >= NMELT) return;
    int n = (int)(i >> 7), d = (int)(i & 127), b = n >> 8;
    float g  = 0.1f * tanhf(gb[(long)b * 256 + d]);
    float be = 0.1f * tanhf(gb[(long)b * 256 + 128 + d]);
    nodes[i] = (1.f + g) * (x3[i] + inj[i]) + be;
}

// h_module attention: one 256-thread block per (b,h); softmax over 256 cells
__global__ void k_hmod_attn(const float* __restrict__ zq, const float* __restrict__ k,
                            const float* __restrict__ v, float* __restrict__ ctx) {
    int b = blockIdx.x >> 2, h = blockIdx.x & 3;
    int c = threadIdx.x;
    __shared__ float red[8];
    __shared__ float smax, ssum;
    __shared__ float lctx[32];
    long row = ((long)(b * 256 + c)) * 128 + h * 32;
    const float* qp = zq + (long)b * 128 + h * 32;
    float dot = 0.f;
#pragma unroll 8
    for (int f = 0; f < 32; ++f) dot += qp[f] * k[row + f];
    dot *= 0.17677669529663687f;
    float m = dot;
#pragma unroll
    for (int off = 16; off; off >>= 1) m = fmaxf(m, __shfl_down(m, off));
    if ((c & 31) == 0) red[c >> 5] = m;
    __syncthreads();
    if (c == 0) {
        float mm = red[0];
        for (int i = 1; i < 8; ++i) mm = fmaxf(mm, red[i]);
        smax = mm;
    }
    __syncthreads();
    float exv = expf(dot - smax);
    float sv  = exv;
#pragma unroll
    for (int off = 16; off; off >>= 1) sv += __shfl_down(sv, off);
    __syncthreads();
    if ((c & 31) == 0) red[c >> 5] = sv;
    if (c < 32) lctx[c] = 0.f;
    __syncthreads();
    if (c == 0) {
        float t = 0.f;
        for (int i = 0; i < 8; ++i) t += red[i];
        ssum = t;
    }
    __syncthreads();
    float w = exv / ssum;
#pragma unroll 8
    for (int f = 0; f < 32; ++f) atomicAdd(&lctx[f], w * v[row + f]);
    __syncthreads();
    if (c < 32) ctx[(long)b * 128 + h * 32 + c] = lctx[c];
}

__global__ void k_concat(const float* __restrict__ ctx, const float* __restrict__ z,
                         float* __restrict__ zin) {
    int i = blockIdx.x * 256 + threadIdx.x;
    if (i >= 32 * 256) return;
    int b = i >> 8, d = i & 255;
    zin[i] = (d < 128) ? ctx[b * 128 + d] : z[b * 128 + (d - 128)];
}

__global__ void k_zup(const float* __restrict__ z, const float* __restrict__ m,
                      const float* __restrict__ sc, float* __restrict__ out) {
    int n = blockIdx.x, d = threadIdx.x;
    float v  = z[n * 128 + d] + m[n * 128 + d];
    float ss = blockSum128(v * v);
    out[n * 128 + d] = sc[d] * v * rsqrtf(ss * (1.f / 128.f) + 1e-6f);
}

__global__ void k_cat(const float* __restrict__ nodes, const float* __restrict__ z,
                      float* __restrict__ cat) {
    long i = (long)blockIdx.x * 256 + threadIdx.x;
    if (i >= (long)NNODE * 256) return;
    int n = (int)(i >> 8), d = (int)(i & 255);
    cat[i] = (d < 128) ? nodes[(long)n * 128 + d] : z[(long)(n >> 8) * 128 + (d - 128)];
}

__global__ void k_final(const float* __restrict__ h, const float* __restrict__ w,
                        const float* __restrict__ b, float* __restrict__ out) {
    int n = blockIdx.x, d = threadIdx.x;
    float s = blockSum128(h[(long)n * 128 + d] * w[d]);
    if (d == 0) out[n] = s + b[0];
}

// ---------------------------------------------------------------------------
extern "C" void kernel_launch(void* const* d_in, const int* in_sizes, int n_in,
                              void* d_out, int out_size, void* d_ws, size_t ws_size,
                              hipStream_t stream) {
    const float* x_cell = (const float*)d_in[0];
    const int* ei[3] = {(const int*)d_in[1], (const int*)d_in[2], (const int*)d_in[3]};
    int E[3] = {in_sizes[1] / 2, in_sizes[2] / 2, in_sizes[3] / 2};

    const float* P[64];
    for (int i = 5; i < n_in && i < 64; ++i) P[i] = (const float*)d_in[i];
    const float* embW = P[5];  const float* embB = P[6];  const float* embS = P[7];
    const float* g1W = P[8];   const float* g1as = P[9];  const float* g1ad = P[10]; const float* g1b = P[11];
    const float* g2W = P[12];  const float* g2as = P[13]; const float* g2ad = P[14]; const float* g2b = P[15];
    const float* norm1 = P[16]; const float* norm2 = P[17]; const float* normH = P[18];
    const float* hWq = P[19]; const float* hbq = P[20];
    const float* hWk = P[21]; const float* hbk = P[22];
    const float* hWv = P[23]; const float* hbv = P[24];
    const float* krel = P[25]; const float* vrel = P[26]; const float* prel = P[27];
    const float* hWo = P[28]; const float* hbo = P[29]; const float* hskip = P[30];
    const float* injW = P[31]; const float* injB = P[32];
    const float* fW1 = P[33]; const float* fb1 = P[34]; const float* fW2 = P[35]; const float* fb2 = P[36];
    const float* mWq = P[37]; const float* mbq = P[38];
    const float* mWk = P[39]; const float* mbk = P[40];
    const float* mWv = P[41]; const float* mbv = P[42];
    const float* mW1 = P[43]; const float* mb1 = P[44]; const float* mW2 = P[45]; const float* mb2 = P[46];
    const float* hsc = P[47]; const float* z0 = P[48];
    const float* roW1 = P[49]; const float* rob1 = P[50]; const float* roW2 = P[51]; const float* rob2 = P[52];

    // workspace layout
    float* ws = (float*)d_ws;
    long o = 0;
    auto al = [&](long n) { float* p = ws + o; o += n; return p; };
    float* emb   = al(NMELT);
    float* inj   = al(NMELT);
    float* nodes = al(NMELT);
    float* xA    = al(NMELT);      // x1 / hgt q / x3 / hmod k / readout h
    float* xB    = al(NMELT);      // x2 / hmod v
    float* xC    = al(NMELT);      // hgt out
    float* h3    = al(3L * NMELT); // GAT h per type; HGT kp per type; readout cat
    float* v3    = al(3L * NMELT); // HGT vp per type
    float* acc   = al(NMELT);      // GAT accumulator; HGT agg
    float* asb   = al(3L * NNODE * 2);
    float* adb   = al(3L * NNODE * 2);
    float* mg    = al(3L * NNODE * 2);   // uint-encoded max
    float* sg    = al(3L * NNODE * 2);
    long exTot = (long)(E[0] + E[1] + E[2] + 3 * NNODE) * 2;
    float* exg   = al(exTot);
    float* mh    = al((long)NNODE * 4);  // uint-encoded max
    float* sh    = al((long)NNODE * 4);
    long ET = (long)E[0] + E[1] + E[2];
    float* sch   = al(ET * 4);
    float* exh   = al(ET * 4);
    float* gb    = al(32 * 256);
    float* t1    = al(32 * 256);
    float* zq    = al(32 * 128);
    float* ctx   = al(32 * 128);
    float* zin   = al(32 * 256);
    float* m1    = al(32 * 128);
    float* m2    = al(32 * 128);
    float* za    = al(32 * 128);
    float* zbuf  = al(32 * 128);
    (void)ws_size;

    auto gemm = [&](const float* A, const float* Bw, const float* bias, float* Cc,
                    int M, int Nn, int K, int lda, int ldc, int act,
                    int batch, long aB, long bB, long cB) {
        dim3 g(Nn / 32, (M + 63) / 64, batch);
        k_gemm<<<g, 128, 0, stream>>>(A, Bw, bias, Cc, M, Nn, K, lda, Nn, ldc,
                                      aB, bB, cB, act);
    };

    const unsigned ORD_NEG_INF = 0x007FFFFFu;

    // one GAT super-layer (sum over 3 edge types)
    auto gat_layer = [&](const float* xin, const float* Wb, const float* asB,
                         const float* adB, const float* bB, const float* nrm, float* xout) {
        gemm(xin, Wb, nullptr, h3, NNODE, 128, 128, 128, 128, 0, 3, 0, 128 * 128, NMELT);
        k_gat_coef<<<dim3(NNODE, 3), 128, 0, stream>>>(h3, asB, adB, asb, adb);
        k_fill_u32<<<(3 * NNODE * 2 + 255) / 256, 256, 0, stream>>>((unsigned*)mg, 3L * NNODE * 2, ORD_NEG_INF);
        k_fill_f32<<<(3 * NNODE * 2 + 255) / 256, 256, 0, stream>>>(sg, 3L * NNODE * 2, 0.f);
        k_accinit3<<<(NMELT + 255) / 256, 256, 0, stream>>>(acc, bB, bB + 128, bB + 256);
        long exOff = 0;
        for (int i = 0; i < 3; ++i) {
            int tot = E[i] + NNODE;
            const float* as_ = asb + (long)i * NNODE * 2;
            const float* ad_ = adb + (long)i * NNODE * 2;
            unsigned* me = (unsigned*)mg + (long)i * NNODE * 2;
            float* se = sg + (long)i * NNODE * 2;
            k_gat_eA<<<(tot + 255) / 256, 256, 0, stream>>>(ei[i], E[i], as_, ad_, me);
            k_gat_eB<<<(tot + 255) / 256, 256, 0, stream>>>(ei[i], E[i], as_, ad_, me, exg + exOff, se);
            k_gat_eC<<<(tot + 7) / 8, 256, 0, stream>>>(ei[i], E[i], h3 + (long)i * NMELT, exg + exOff, se, acc);
            exOff += (long)tot * 2;
        }
        k_combine<<<NNODE, 128, 0, stream>>>(xin, acc, nrm, xout);
    };

    // ---- one-time precompute ----
    k_embed<<<NNODE, 128, 0, stream>>>(x_cell, embW, embB, embS, emb);
    gemm(emb, injW, injB, inj, NNODE, 128, 128, 128, 128, 0, 1, 0, 0, 0);
    hipMemcpyAsync(nodes, emb, (size_t)NMELT * sizeof(float), hipMemcpyDeviceToDevice, stream);
    k_rowbcast<<<(32 * 128 + 255) / 256, 256, 0, stream>>>(za, z0, 32 * 128);
    float* zcur = za;
    float* znext = zbuf;

    long hgtOff[3] = {0, (long)E[0], (long)E[0] + E[1]};

    for (int cyc = 0; cyc < 3; ++cyc) {
        for (int t = 0; t < 2; ++t) {
            // GAT layer 1: nodes -> xA ; GAT layer 2: xA -> xB
            gat_layer(nodes, g1W, g1as, g1ad, g1b, norm1, xA);
            gat_layer(xA, g2W, g2as, g2ad, g2b, norm2, xB);

            // ---- HGT on xB ----
            gemm(xB, hWq, hbq, xA, NNODE, 128, 128, 128, 128, 0, 1, 0, 0, 0);      // q -> xA
            gemm(xB, hWk, hbk, nodes, NNODE, 128, 128, 128, 128, 0, 1, 0, 0, 0);   // k -> nodes
            for (int i = 0; i < 3; ++i)                                            // kp -> h3
                gemm(nodes, krel + (long)i * 4096, nullptr, h3 + (long)i * NMELT,
                     NNODE, 32, 32, 128, 128, 0, 4, 32, 1024, 32);
            gemm(xB, hWv, hbv, nodes, NNODE, 128, 128, 128, 128, 0, 1, 0, 0, 0);   // v -> nodes
            for (int i = 0; i < 3; ++i)                                            // vp -> v3
                gemm(nodes, vrel + (long)i * 4096, nullptr, v3 + (long)i * NMELT,
                     NNODE, 32, 32, 128, 128, 0, 4, 32, 1024, 32);
            k_fill_u32<<<(NNODE * 4 + 255) / 256, 256, 0, stream>>>((unsigned*)mh, (long)NNODE * 4, ORD_NEG_INF);
            k_fill_f32<<<(NNODE * 4 + 255) / 256, 256, 0, stream>>>(sh, (long)NNODE * 4, 0.f);
            for (int i = 0; i < 3; ++i)
                k_hgt_score<<<(E[i] + 255) / 256, 256, 0, stream>>>(
                    ei[i], E[i], xA, h3 + (long)i * NMELT, prel + i * 4,
                    sch + hgtOff[i] * 4, (unsigned*)mh);
            for (int i = 0; i < 3; ++i)
                k_hgt_exp<<<(E[i] + 255) / 256, 256, 0, stream>>>(
                    ei[i], E[i], sch + hgtOff[i] * 4, (unsigned*)mh, exh + hgtOff[i] * 4, sh);
            k_fill_f32<<<(NMELT + 255) / 256, 256, 0, stream>>>(acc, NMELT, 0.f);
            for (int i = 0; i < 3; ++i)
                k_hgt_agg<<<(E[i] + 7) / 8, 256, 0, stream>>>(
                    ei[i], E[i], v3 + (long)i * NMELT, exh + hgtOff[i] * 4, sh, acc);
            k_gelu_ip<<<(NMELT + 255) / 256, 256, 0, stream>>>(acc, NMELT);
            gemm(acc, hWo, hbo, xC, NNODE, 128, 128, 128, 128, 0, 1, 0, 0, 0);
            k_mixcomb<<<NNODE, 128, 0, stream>>>(xB, xC, hskip, normH, xA);         // x3 -> xA

            // ---- FiLM ----
            gemm(zcur, fW1, fb1, t1, 32, 256, 128, 128, 256, 1, 1, 0, 0, 0);
            gemm(t1, fW2, fb2, gb, 32, 256, 256, 256, 256, 0, 1, 0, 0, 0);
            k_film<<<(NMELT + 255) / 256, 256, 0, stream>>>(xA, inj, gb, nodes);
        }
        // ---- h_module ----
        gemm(zcur, mWq, mbq, zq, 32, 128, 128, 128, 128, 0, 1, 0, 0, 0);
        gemm(nodes, mWk, mbk, xA, NNODE, 128, 128, 128, 128, 0, 1, 0, 0, 0);
        gemm(nodes, mWv, mbv, xB, NNODE, 128, 128, 128, 128, 0, 1, 0, 0, 0);
        k_hmod_attn<<<128, 256, 0, stream>>>(zq, xA, xB, ctx);
        k_concat<<<(32 * 256 + 255) / 256, 256, 0, stream>>>(ctx, zcur, zin);
        gemm(zin, mW1, mb1, m1, 32, 128, 256, 256, 128, 1, 1, 0, 0, 0);
        gemm(m1, mW2, mb2, m2, 32, 128, 128, 128, 128, 0, 1, 0, 0, 0);
        k_zup<<<32, 128, 0, stream>>>(zcur, m2, hsc, znext);
        float* tmp = zcur; zcur = znext; znext = tmp;
    }

    // ---- readout ----
    float* cat = h3;  // reuse (needs 2*NMELT of 3*NMELT)
    k_cat<<<((long)NNODE * 256 + 255) / 256, 256, 0, stream>>>(nodes, zcur, cat);
    gemm(cat, roW1, rob1, xA, NNODE, 128, 256, 256, 128, 1, 1, 0, 0, 0);
    k_final<<<NNODE, 128, 0, stream>>>(xA, roW2, rob2, (float*)d_out);
}
